// DeformableDecoderLayer_73143293051452
// MI455X (gfx1250) — matrix operand order repaired and checked by reading, not compiled, not run on hardware
//
#include <hip/hip_runtime.h>
#include <math.h>

// ---------------- constants (from the reference) ----------------
#define B_   16
#define LQ_  900
#define D_   256
#define NH_  8
#define HD_  32
#define NL_  4
#define NP_  4
#define DFF_ 1024
#define S_   13294

typedef __bf16 bf16_t;
typedef __attribute__((ext_vector_type(16))) __bf16 v16bf;
typedef __attribute__((ext_vector_type(8)))  __bf16 v8bf;
typedef __attribute__((ext_vector_type(8)))  float  v8f;

static __device__ __forceinline__ v8bf zero8() {
  v8bf z;
#pragma unroll
  for (int i = 0; i < 8; i++) z[i] = (__bf16)0.0f;
  return z;
}
static __device__ __forceinline__ v8f zerov8f() {
  v8f z;
#pragma unroll
  for (int i = 0; i < 8; i++) z[i] = 0.0f;
  return z;
}
static __device__ __forceinline__ v16bf pack16(v8bf lo, v8bf hi) {
  v16bf r;
#pragma unroll
  for (int i = 0; i < 8; i++) { r[i] = lo[i]; r[i + 8] = hi[i]; }
  return r;
}

// ---- gfx1250 async global->LDS copy (ASYNCcnt path, bypasses VGPRs) ----
// VDST vgpr carries the LDS byte offset; GV addressing (64-bit vaddr, saddr=off).
// Low 32 bits of a generic pointer to __shared__ are the LDS offset on AMDGPU.
static __device__ __forceinline__ void async_cp16(void* lds, const void* gptr) {
  unsigned ldsoff = (unsigned)(unsigned long long)lds;
  asm volatile("global_load_async_to_lds_b128 %0, %1, off"
               :: "v"(ldsoff), "v"(gptr) : "memory");
}
static __device__ __forceinline__ void async_wait0() {
  asm volatile("s_wait_asynccnt 0" ::: "memory");
}

// =================================================================
// Generic WMMA GEMM:  out = act( A @ Wt^T + bias ) (+resid) (+mask)
//   A: M x K  (f32 OR bf16, row-major)
//   Wt: N x K bf16 (pre-transposed weights)
//   Block 256 threads = 8 waves; tile 128x128, BK=32.
//   Wave grid 2(M) x 4(N); per-wave 64x32 => 4x2 wmma tiles.
//   Tile staging uses GLOBAL_LOAD_ASYNC_TO_LDS_B128 for bf16 sources.
// =================================================================
__global__ __launch_bounds__(256) void gemm_wmma(
    int M, int N, int K,
    const float* __restrict__ Af, const bf16_t* __restrict__ Ab,
    const bf16_t* __restrict__ Wt,
    const float* __restrict__ bias,
    const float* __restrict__ resid,
    const unsigned char* __restrict__ mask,
    float* __restrict__ outF, bf16_t* __restrict__ outB, int relu)
{
  __shared__ __align__(16) bf16_t As[128][40];
  __shared__ __align__(16) bf16_t Bs[128][40];

  const int tid  = threadIdx.x;
  const int lane = tid & 31;
  const int wave = tid >> 5;
  const int wm   = wave >> 2;      // 0..1
  const int wn   = wave & 3;       // 0..3
  const int bm   = blockIdx.x * 128;
  const int bn   = blockIdx.y * 128;

  v8f acc[4][2];
#pragma unroll
  for (int i = 0; i < 4; i++)
#pragma unroll
    for (int j = 0; j < 2; j++) acc[i][j] = zerov8f();

  const int r  = tid >> 1;          // 0..127 : tile row (A) / tile col (B)
  const int hc = (tid & 1) * 16;    // 0 or 16 : K half within BK=32

  for (int kk = 0; kk < K; kk += 32) {
    // ---- stage A tile (128 x 32) ----
    {
      const int grow = bm + r;
      if (Ab) {
        if (grow < M) {
          const bf16_t* p = Ab + (size_t)grow * K + kk + hc;
          async_cp16(&As[r][hc],     p);
          async_cp16(&As[r][hc + 8], p + 8);
        } else {
          *(v8bf*)&As[r][hc]     = zero8();
          *(v8bf*)&As[r][hc + 8] = zero8();
        }
      } else {
        if (grow < M) {
          const float4* p = (const float4*)(Af + (size_t)grow * K + kk + hc);
          float4 f0 = p[0], f1 = p[1], f2 = p[2], f3 = p[3];
          bf16_t* d = &As[r][hc];
          d[0]=(bf16_t)f0.x; d[1]=(bf16_t)f0.y; d[2]=(bf16_t)f0.z; d[3]=(bf16_t)f0.w;
          d[4]=(bf16_t)f1.x; d[5]=(bf16_t)f1.y; d[6]=(bf16_t)f1.z; d[7]=(bf16_t)f1.w;
          d[8]=(bf16_t)f2.x; d[9]=(bf16_t)f2.y; d[10]=(bf16_t)f2.z; d[11]=(bf16_t)f2.w;
          d[12]=(bf16_t)f3.x; d[13]=(bf16_t)f3.y; d[14]=(bf16_t)f3.z; d[15]=(bf16_t)f3.w;
          if (kk + 32 < K) __builtin_prefetch(Af + (size_t)grow * K + kk + 32 + hc, 0, 3);
        } else {
          *(v8bf*)&As[r][hc]     = zero8();
          *(v8bf*)&As[r][hc + 8] = zero8();
        }
      }
    }
    // ---- stage B tile (128 x 32) from Wt (N x K) : async copy ----
    {
      const bf16_t* p = Wt + (size_t)(bn + r) * K + kk + hc;
      async_cp16(&Bs[r][hc],     p);
      async_cp16(&Bs[r][hc + 8], p + 8);
    }
    async_wait0();
    __syncthreads();

    // ---- fragments + 8 WMMAs ----
    v16bf afr[4], bfr[2];
    const int k0a = 8 * (lane >> 4);
#pragma unroll
    for (int tm = 0; tm < 4; tm++) {
      const int row = wm * 64 + tm * 16 + (lane & 15);
      v8bf lo = *(const v8bf*)&As[row][k0a];
      v8bf hi = *(const v8bf*)&As[row][k0a + 16];
      afr[tm] = pack16(lo, hi);
    }
    const int k0b = 8 * (lane >> 4);
#pragma unroll
    for (int tn = 0; tn < 2; tn++) {
      const int col = wn * 32 + tn * 16 + (lane & 15);
      v8bf lo = *(const v8bf*)&Bs[col][k0b];
      v8bf hi = *(const v8bf*)&Bs[col][k0b + 16];
      bfr[tn] = pack16(lo, hi);
    }
#pragma unroll
    for (int tm = 0; tm < 4; tm++)
#pragma unroll
      for (int tn = 0; tn < 2; tn++)
        acc[tm][tn] = __builtin_amdgcn_wmma_f32_16x16x32_bf16(
            false, afr[tm], false, bfr[tn], (short)0, acc[tm][tn], false, false);
    __syncthreads();
  }

  // ---- epilogue ----
#pragma unroll
  for (int tm = 0; tm < 4; tm++) {
    const int rowb = bm + wm * 64 + tm * 16 + 8 * (lane >> 4);
#pragma unroll
    for (int tn = 0; tn < 2; tn++) {
      const int col = bn + wn * 32 + tn * 16 + (lane & 15);
      const float bv = bias ? bias[col] : 0.0f;
#pragma unroll
      for (int v = 0; v < 8; v++) {
        const int row = rowb + v;
        if (row < M) {
          float x = acc[tm][tn][v] + bv;
          if (relu) x = fmaxf(x, 0.0f);
          if (resid) x += resid[(size_t)row * N + col];
          if (mask && mask[row]) x = 0.0f;
          if (outF) outF[(size_t)row * N + col] = x;
          if (outB) outB[(size_t)row * N + col] = (bf16_t)x;
        }
      }
    }
  }
}

// =================================================================
// Flash attention (LQ=900, HD=32). Block 128 = 4 waves; each wave
// owns 16 query rows; keys streamed in chunks of 32.
// qh/kh/vh/out layout: (B, LQ, NH, HD) bf16.
// =================================================================
__global__ __launch_bounds__(128) void attn_kernel(
    const bf16_t* __restrict__ qh, const bf16_t* __restrict__ kh,
    const bf16_t* __restrict__ vh, bf16_t* __restrict__ attn_out)
{
  __shared__ __align__(16) bf16_t Vt[32][40];       // [d][key]
  __shared__ __align__(16) bf16_t Ps[4][16][40];    // per-wave P tile

  const int tid = threadIdx.x, lane = tid & 31, wave = tid >> 5;
  const int b = blockIdx.z, h = blockIdx.y;
  const int q0 = blockIdx.x * 64 + wave * 16;
  const float scale = 0.17677669529663687f;         // 1/sqrt(32)

  // Q fragment (A layout): lane row = q0 + (lane&15), K split by half-group
  v16bf qfr;
  {
    const int q  = q0 + (lane & 15);
    const int d0 = 8 * (lane >> 4);
    v8bf lo = zero8(), hi = zero8();
    if (q < LQ_) {
      const bf16_t* p = qh + (((size_t)b * LQ_ + q) * NH_ + h) * HD_;
      lo = *(const v8bf*)(p + d0);
      hi = *(const v8bf*)(p + d0 + 16);
    }
    qfr = pack16(lo, hi);
  }

  float rowmax[8], rowsum[8];
  v8f o0 = zerov8f(), o1 = zerov8f();
#pragma unroll
  for (int v = 0; v < 8; v++) { rowmax[v] = -3.0e30f; rowsum[v] = 0.0f; }

  const int NCH = (LQ_ + 31) / 32;   // 29
  for (int c = 0; c < NCH; c++) {
    const int key0 = c * 32;
    // ---- stage V chunk transposed: Vt[d][key] ----
    {
      const int j  = key0 + (tid >> 2);
      const int dc = (tid & 3) * 8;
      v8bf vv = zero8();
      if (j < LQ_) vv = *(const v8bf*)(vh + (((size_t)b * LQ_ + j) * NH_ + h) * HD_ + dc);
#pragma unroll
      for (int i = 0; i < 8; i++) Vt[dc + i][tid >> 2] = vv[i];
    }
    __syncthreads();

    // ---- S = Q * K^T (2 WMMAs covering 32 keys) ----
    v8f s[2];
#pragma unroll
    for (int tn = 0; tn < 2; tn++) {
      const int j = key0 + tn * 16 + (lane & 15);
      v8bf lo = zero8(), hi = zero8();
      if (j < LQ_) {
        const bf16_t* p = kh + (((size_t)b * LQ_ + j) * NH_ + h) * HD_ + 8 * (lane >> 4);
        lo = *(const v8bf*)p;
        hi = *(const v8bf*)(p + 16);
      }
      v16bf kfr = pack16(lo, hi);
      s[tn] = __builtin_amdgcn_wmma_f32_16x16x32_bf16(
          false, qfr, false, kfr, (short)0, zerov8f(), false, false);
    }
    // scale + key mask
#pragma unroll
    for (int tn = 0; tn < 2; tn++) {
      const int j = key0 + tn * 16 + (lane & 15);
#pragma unroll
      for (int v = 0; v < 8; v++) {
        float x = s[tn][v] * scale;
        if (j >= LQ_) x = -3.0e30f;
        s[tn][v] = x;
      }
    }
    // ---- online softmax (row reductions within 16-lane half-groups) ----
#pragma unroll
    for (int v = 0; v < 8; v++) {
      float t = fmaxf(s[0][v], s[1][v]);
#pragma unroll
      for (int o = 1; o < 16; o <<= 1) t = fmaxf(t, __shfl_xor(t, o, 32));
      const float mnew = fmaxf(rowmax[v], t);
      const float alpha = __expf(rowmax[v] - mnew);
      rowmax[v] = mnew;
      float p0 = __expf(s[0][v] - mnew);
      float p1 = __expf(s[1][v] - mnew);
      s[0][v] = p0; s[1][v] = p1;
      float ps = p0 + p1;
#pragma unroll
      for (int o = 1; o < 16; o <<= 1) ps += __shfl_xor(ps, o, 32);
      rowsum[v] = rowsum[v] * alpha + ps;
      o0[v] *= alpha;
      o1[v] *= alpha;
    }
    // ---- P (C layout) -> LDS -> A layout ----
    {
      const int rr = 8 * (lane >> 4), cc = lane & 15;
#pragma unroll
      for (int v = 0; v < 8; v++) {
        Ps[wave][rr + v][cc]      = (bf16_t)s[0][v];
        Ps[wave][rr + v][16 + cc] = (bf16_t)s[1][v];
      }
    }
    __syncthreads();
    v16bf pfr;
    {
      const int m  = lane & 15;
      const int k0 = 8 * (lane >> 4);
      v8bf lo = *(const v8bf*)&Ps[wave][m][k0];
      v8bf hi = *(const v8bf*)&Ps[wave][m][k0 + 16];
      pfr = pack16(lo, hi);
    }
    // ---- O += P * V (2 WMMAs over HD=32) ----
#pragma unroll
    for (int tn = 0; tn < 2; tn++) {
      const int d  = tn * 16 + (lane & 15);
      const int k0 = 8 * (lane >> 4);
      v8bf lo = *(const v8bf*)&Vt[d][k0];
      v8bf hi = *(const v8bf*)&Vt[d][k0 + 16];
      v16bf vfr = pack16(lo, hi);
      if (tn == 0)
        o0 = __builtin_amdgcn_wmma_f32_16x16x32_bf16(false, pfr, false, vfr, (short)0, o0, false, false);
      else
        o1 = __builtin_amdgcn_wmma_f32_16x16x32_bf16(false, pfr, false, vfr, (short)0, o1, false, false);
    }
    __syncthreads();
  }

  // ---- normalize + store ----
  {
    const int rr = 8 * (lane >> 4), d = lane & 15;
#pragma unroll
    for (int v = 0; v < 8; v++) {
      const int q = q0 + rr + v;
      if (q < LQ_) {
        const float inv = (rowsum[v] > 0.0f) ? 1.0f / rowsum[v] : 0.0f;
        bf16_t* p = attn_out + (((size_t)b * LQ_ + q) * NH_ + h) * HD_;
        p[d]      = (bf16_t)(o0[v] * inv);
        p[16 + d] = (bf16_t)(o1[v] * inv);
      }
    }
  }
}

// =================================================================
// LayerNorm over D=256, wave per row. Optional bf16 copy and
// bf16(y + pos) copy.
// =================================================================
__global__ __launch_bounds__(256) void ln_kernel(
    const float* __restrict__ x, const float* __restrict__ g,
    const float* __restrict__ be, const float* __restrict__ pos,
    float* __restrict__ outF, bf16_t* __restrict__ outB,
    bf16_t* __restrict__ outBpos, int M)
{
  const int row  = blockIdx.x * 8 + (threadIdx.x >> 5);
  const int lane = threadIdx.x & 31;
  if (row >= M) return;
  const float* xr = x + (size_t)row * D_ + lane * 8;
  float v[8];
  {
    float4 a = ((const float4*)xr)[0], b = ((const float4*)xr)[1];
    v[0]=a.x; v[1]=a.y; v[2]=a.z; v[3]=a.w; v[4]=b.x; v[5]=b.y; v[6]=b.z; v[7]=b.w;
  }
  float s = 0.0f;
#pragma unroll
  for (int i = 0; i < 8; i++) s += v[i];
#pragma unroll
  for (int o = 1; o < 32; o <<= 1) s += __shfl_xor(s, o, 32);
  const float mean = s * (1.0f / 256.0f);
  float q = 0.0f;
#pragma unroll
  for (int i = 0; i < 8; i++) { float d = v[i] - mean; q += d * d; }
#pragma unroll
  for (int o = 1; o < 32; o <<= 1) q += __shfl_xor(q, o, 32);
  const float rstd = rsqrtf(q * (1.0f / 256.0f) + 1e-5f);
#pragma unroll
  for (int i = 0; i < 8; i++) {
    const int c = lane * 8 + i;
    const float y = (v[i] - mean) * rstd * g[c] + be[c];
    const size_t idx = (size_t)row * D_ + c;
    if (outF) outF[idx] = y;
    if (outB) outB[idx] = (bf16_t)y;
    if (outBpos) outBpos[idx] = (bf16_t)(y + pos[idx]);
  }
}

// elementwise: out = bf16(a + p)
__global__ __launch_bounds__(256) void add_cast_kernel(
    const float* __restrict__ a, const float* __restrict__ p,
    bf16_t* __restrict__ out, int n)
{
  const int i = blockIdx.x * 256 + threadIdx.x;
  if (i < n) out[i] = (bf16_t)(a[i] + p[i]);
}

// weight transpose + cast: w (K x N f32) -> wt (N x K bf16)
__global__ __launch_bounds__(256) void wtrans_kernel(
    const float* __restrict__ w, bf16_t* __restrict__ wt, int K, int N)
{
  const int i = blockIdx.x * 256 + threadIdx.x;
  if (i < K * N) {
    const int k = i / N, n = i % N;
    wt[(size_t)n * K + k] = (bf16_t)w[i];
  }
}

// =================================================================
// Multi-scale deformable sampling. One wave per (b,q,h), lane = d.
// value: (B, S, NH, HD) bf16.  samp out: (B, LQ, D) bf16.
// =================================================================
__global__ __launch_bounds__(256) void msda_kernel(
    const float* __restrict__ ref, const float* __restrict__ off,
    const float* __restrict__ awl, const bf16_t* __restrict__ value,
    bf16_t* __restrict__ samp)
{
  const int wid  = blockIdx.x * 8 + (threadIdx.x >> 5);
  const int lane = threadIdx.x & 31;          // channel d
  if (wid >= B_ * LQ_ * NH_) return;
  const int h = wid % NH_;
  const int t = wid / NH_;
  const int q = t % LQ_;
  const int b = t / LQ_;

  // softmax over the 16 attention logits of this (b,q,h)
  const float* wl = awl + ((size_t)b * LQ_ + q) * (NH_ * NL_ * NP_) + h * (NL_ * NP_);
  float w[16], mx = -3.0e30f;
#pragma unroll
  for (int i = 0; i < 16; i++) { w[i] = wl[i]; mx = fmaxf(mx, w[i]); }
  float sm = 0.0f;
#pragma unroll
  for (int i = 0; i < 16; i++) { w[i] = __expf(w[i] - mx); sm += w[i]; }
  const float inv = 1.0f / sm;

  const int Hs[4]   = {100, 50, 25, 13};
  const int Ws[4]   = {100, 50, 25, 13};
  const int LSIs[4] = {0, 10000, 12500, 13125};

  const float* rp = ref + (((size_t)b * LQ_ + q) * NL_) * 2;
  const float* op = off + ((size_t)b * LQ_ + q) * (NH_ * NL_ * NP_ * 2) + h * (NL_ * NP_ * 2);
  float acc = 0.0f;
#pragma unroll
  for (int l = 0; l < NL_; l++) {
    const int Hc = Hs[l], Wc = Ws[l];
    const float fW = (float)Wc, fH = (float)Hc;
    const float rx = rp[l * 2 + 0], ry = rp[l * 2 + 1];
    const bf16_t* vbase = value + (((size_t)b * S_ + LSIs[l]) * NH_ + h) * HD_ + lane;
#pragma unroll
    for (int p = 0; p < NP_; p++) {
      const float ox = op[(l * 4 + p) * 2 + 0], oy = op[(l * 4 + p) * 2 + 1];
      const float x = rx * fW + ox - 0.5f;          // (ref + off/norm)*W - 0.5
      const float y = ry * fH + oy - 0.5f;
      const float x0 = floorf(x), y0 = floorf(y);
      const float lx = x - x0, ly = y - y0;
      const int xi = (int)x0, yi = (int)y0;
      float sacc = 0.0f;
#pragma unroll
      for (int dy = 0; dy < 2; dy++)
#pragma unroll
        for (int dx = 0; dx < 2; dx++) {
          const int xx = xi + dx, yy = yi + dy;
          const float wgt = (dy ? ly : 1.0f - ly) * (dx ? lx : 1.0f - lx);
          if (xx >= 0 && xx < Wc && yy >= 0 && yy < Hc) {
            const float gv = (float)vbase[(size_t)(yy * Wc + xx) * (NH_ * HD_)];
            sacc += wgt * gv;
          }
        }
      acc += w[l * 4 + p] * inv * sacc;
    }
  }
  samp[((size_t)b * LQ_ + q) * D_ + h * HD_ + lane] = (bf16_t)acc;
}

// =================================================================
// Host-side orchestration
// =================================================================
extern "C" void kernel_launch(void* const* d_in, const int* in_sizes, int n_in,
                              void* d_out, int out_size, void* d_ws, size_t ws_size,
                              hipStream_t stream)
{
  const float* tgt   = (const float*)d_in[0];
  const float* qpos  = (const float*)d_in[1];
  const float* refp  = (const float*)d_in[2];
  const float* src   = (const float*)d_in[3];
  const unsigned char* pmask = (const unsigned char*)d_in[6];
  const float* Wq = (const float*)d_in[7],  *bq = (const float*)d_in[8];
  const float* Wk = (const float*)d_in[9],  *bk = (const float*)d_in[10];
  const float* Wv = (const float*)d_in[11], *bv = (const float*)d_in[12];
  const float* Wo = (const float*)d_in[13], *bo = (const float*)d_in[14];
  const float* g2 = (const float*)d_in[15], *be2 = (const float*)d_in[16];
  const float* Woff = (const float*)d_in[17], *boff = (const float*)d_in[18];
  const float* Waw  = (const float*)d_in[19], *baw  = (const float*)d_in[20];
  const float* Wval = (const float*)d_in[21], *bval = (const float*)d_in[22];
  const float* Wout = (const float*)d_in[23], *bout = (const float*)d_in[24];
  const float* g1 = (const float*)d_in[25], *be1 = (const float*)d_in[26];
  const float* W1f = (const float*)d_in[27], *b1f = (const float*)d_in[28];
  const float* W2f = (const float*)d_in[29], *b2f = (const float*)d_in[30];
  const float* g3 = (const float*)d_in[31], *be3 = (const float*)d_in[32];

  char* ws = (char*)d_ws;
  size_t cur = 0;
  auto alloc = [&](size_t bytes) -> char* {
    char* p = ws + cur;
    cur += (bytes + 255) & ~(size_t)255;
    return p;
  };

  const int    MQ = B_ * LQ_;                 // 14400
  const size_t nA = (size_t)MQ * D_;          // 3,686,400

  bf16_t* actQ = (bf16_t*)alloc(nA * 2);
  bf16_t* qh   = (bf16_t*)alloc(nA * 2);
  bf16_t* khb  = (bf16_t*)alloc(nA * 2);
  bf16_t* vhb  = (bf16_t*)alloc(nA * 2);
  bf16_t* attn = (bf16_t*)alloc(nA * 2);
  bf16_t* q1bf = (bf16_t*)alloc(nA * 2);
  bf16_t* sampb= (bf16_t*)alloc(nA * 2);
  bf16_t* t2bf = (bf16_t*)alloc(nA * 2);
  bf16_t* ffh  = (bf16_t*)alloc((size_t)MQ * DFF_ * 2);
  bf16_t* val  = (bf16_t*)alloc((size_t)B_ * S_ * D_ * 2);
  float*  xtmp = (float*)alloc(nA * 4);
  float*  tgt1 = (float*)alloc(nA * 4);
  float*  tgt2 = (float*)alloc(nA * 4);
  float*  offb = (float*)alloc(nA * 4);
  float*  awl  = (float*)alloc((size_t)MQ * 128 * 4);
  bf16_t* WqT   = (bf16_t*)alloc(65536 * 2);
  bf16_t* WkT   = (bf16_t*)alloc(65536 * 2);
  bf16_t* WvT   = (bf16_t*)alloc(65536 * 2);
  bf16_t* WoT   = (bf16_t*)alloc(65536 * 2);
  bf16_t* WoffT = (bf16_t*)alloc(65536 * 2);
  bf16_t* WawT  = (bf16_t*)alloc(32768 * 2);
  bf16_t* WvalT = (bf16_t*)alloc(65536 * 2);
  bf16_t* WoutT = (bf16_t*)alloc(65536 * 2);
  bf16_t* W1fT  = (bf16_t*)alloc(262144 * 2);
  bf16_t* W2fT  = (bf16_t*)alloc(262144 * 2);

  // ---- weight transpose/cast ----
  wtrans_kernel<<<256,  256, 0, stream>>>(Wq,   WqT,   256, 256);
  wtrans_kernel<<<256,  256, 0, stream>>>(Wk,   WkT,   256, 256);
  wtrans_kernel<<<256,  256, 0, stream>>>(Wv,   WvT,   256, 256);
  wtrans_kernel<<<256,  256, 0, stream>>>(Wo,   WoT,   256, 256);
  wtrans_kernel<<<256,  256, 0, stream>>>(Woff, WoffT, 256, 256);
  wtrans_kernel<<<128,  256, 0, stream>>>(Waw,  WawT,  256, 128);
  wtrans_kernel<<<256,  256, 0, stream>>>(Wval, WvalT, 256, 256);
  wtrans_kernel<<<256,  256, 0, stream>>>(Wout, WoutT, 256, 256);
  wtrans_kernel<<<1024, 256, 0, stream>>>(W1f,  W1fT,  256, 1024);
  wtrans_kernel<<<1024, 256, 0, stream>>>(W2f,  W2fT,  1024, 256);

  auto gemm = [&](int M, int N, int K, const float* Af, const bf16_t* Ab,
                  const bf16_t* Wt, const float* bias, const float* resid,
                  const unsigned char* mask, float* oF, bf16_t* oB, int relu) {
    dim3 grid((M + 127) / 128, N / 128);
    gemm_wmma<<<grid, 256, 0, stream>>>(M, N, K, Af, Ab, Wt, bias, resid, mask, oF, oB, relu);
  };

  // ---- self-attention ----
  add_cast_kernel<<<(int)(nA / 256), 256, 0, stream>>>(tgt, qpos, actQ, (int)nA);
  gemm(MQ, 256, 256, nullptr, actQ, WqT, bq, nullptr, nullptr, nullptr, qh,  0);
  gemm(MQ, 256, 256, nullptr, actQ, WkT, bk, nullptr, nullptr, nullptr, khb, 0);
  gemm(MQ, 256, 256, tgt, nullptr,  WvT, bv, nullptr, nullptr, nullptr, vhb, 0);
  attn_kernel<<<dim3(15, NH_, B_), 128, 0, stream>>>(qh, khb, vhb, attn);
  gemm(MQ, 256, 256, nullptr, attn, WoT, bo, tgt, nullptr, xtmp, nullptr, 0);
  ln_kernel<<<(MQ + 7) / 8, 256, 0, stream>>>(xtmp, g2, be2, qpos, tgt1, nullptr, q1bf, MQ);

  // ---- deformable cross-attention ----
  gemm(B_ * S_, 256, 256, src, nullptr, WvalT, bval, nullptr, pmask, nullptr, val, 0);
  gemm(MQ, 256, 256, nullptr, q1bf, WoffT, boff, nullptr, nullptr, offb, nullptr, 0);
  gemm(MQ, 128, 256, nullptr, q1bf, WawT,  baw,  nullptr, nullptr, awl,  nullptr, 0);
  msda_kernel<<<(B_ * LQ_ * NH_) / 8, 256, 0, stream>>>(refp, offb, awl, val, sampb);
  gemm(MQ, 256, 256, nullptr, sampb, WoutT, bout, tgt1, nullptr, xtmp, nullptr, 0);
  ln_kernel<<<(MQ + 7) / 8, 256, 0, stream>>>(xtmp, g1, be1, nullptr, tgt2, t2bf, nullptr, MQ);

  // ---- FFN ----
  gemm(MQ, 1024, 256, nullptr, t2bf, W1fT, b1f, nullptr, nullptr, nullptr, ffh, 1);
  gemm(MQ, 256, 1024, nullptr, ffh,  W2fT, b2f, tgt2,    nullptr, xtmp, nullptr, 0);
  ln_kernel<<<(MQ + 7) / 8, 256, 0, stream>>>(xtmp, g3, be3, nullptr, (float*)d_out, nullptr, nullptr, MQ);

  (void)in_sizes; (void)n_in; (void)out_size; (void)ws_size;
}
